// Net_22909355557116
// MI455X (gfx1250) — matrix operand order repaired and checked by reading, not compiled, not run on hardware
//
#include <hip/hip_runtime.h>
#include <math.h>

typedef __attribute__((ext_vector_type(16))) _Float16 v16h;
typedef __attribute__((ext_vector_type(8)))  float    v8f;

#define BATCH 2

// ---------------------------------------------------------------------------
// Weight prep: f32 OIHW -> f16 fragment-ordered [ktile][n][klocal]
//   k = ci*ks*ks + fy*ks + fx ; dst[((k>>5)*Np + n)*32 + (k&31)] so each
//   lane's B fragment (16 halfs, k_local = half*16..half*16+15) is one
//   contiguous 32B-aligned load.
// ---------------------------------------------------------------------------
__global__ void prep_weight_kernel(const float* __restrict__ src,
                                   _Float16* __restrict__ dst,
                                   int K, int Kp, int Np, int Cout) {
  int i = blockIdx.x * blockDim.x + threadIdx.x;
  int tot = Kp * Np;
  if (i >= tot) return;
  int k = i / Np;
  int n = i - k * Np;
  float v = 0.f;
  if (k < K && n < Cout) v = src[(size_t)n * K + k];
  dst[((size_t)(k >> 5) * Np + n) * 32 + (k & 31)] = (_Float16)v;
}

// bn(x) = x*s + t with s = g*rsqrt(v+eps), t = b - m*s
__global__ void prep_bn_kernel(const float* __restrict__ g, const float* __restrict__ b,
                               const float* __restrict__ m, const float* __restrict__ v,
                               int C, float* __restrict__ s, float* __restrict__ t) {
  int i = blockIdx.x * blockDim.x + threadIdx.x;
  if (i >= C) return;
  float sc = g[i] * rsqrtf(v[i] + 1e-5f);
  s[i] = sc;
  t[i] = b[i] - m[i] * sc;
}

// ---------------------------------------------------------------------------
// resample_data: gray (if rgb) + space-to-depth by 4. out: [B][16][Hh][Ww]
// ---------------------------------------------------------------------------
__global__ void resample_kernel(const float* __restrict__ src, int Cin,
                                float* __restrict__ out, int Hh, int Ww) {
  int i = blockIdx.x * blockDim.x + threadIdx.x;
  int tot = BATCH * 16 * Hh * Ww;
  if (i >= tot) return;
  int xw = i % Ww;
  int t1 = i / Ww;
  int yh = t1 % Hh;
  int t2 = t1 / Hh;
  int c = t2 % 16;
  int b = t2 / 16;
  int ii = c >> 2, jj = c & 3;
  int H = Hh * 4, W = Ww * 4;
  int y = yh * 4 + ii, x = xw * 4 + jj;
  float g;
  if (Cin == 3) {
    float c0 = src[(((size_t)b * 3 + 0) * H + y) * W + x];
    float c1 = src[(((size_t)b * 3 + 1) * H + y) * W + x];
    float c2 = src[(((size_t)b * 3 + 2) * H + y) * W + x];
    g = 0.299f * c2 + 0.587f * c1 + 0.114f * c0;
  } else {
    g = src[((size_t)b * H + y) * W + x];
  }
  out[i] = g;
}

// ---------------------------------------------------------------------------
// 2x2 average pool, NCHW. dst at Ho x Wo, src at 2Ho x 2Wo.
// ---------------------------------------------------------------------------
__global__ void avgpool2_kernel(const float* __restrict__ src, float* __restrict__ dst,
                                int C, int Ho, int Wo) {
  int i = blockIdx.x * blockDim.x + threadIdx.x;
  int tot = BATCH * C * Ho * Wo;
  if (i >= tot) return;
  int x = i % Wo;
  int t1 = i / Wo;
  int y = t1 % Ho;
  int t2 = t1 / Ho;
  int c = t2 % C;
  int b = t2 / C;
  int Hi = Ho * 2, Wi = Wo * 2;
  const float* p = src + (((size_t)b * C + c) * Hi + 2 * y) * Wi + 2 * x;
  dst[i] = 0.25f * (p[0] + p[1] + p[Wi] + p[Wi + 1]);
}

// ---------------------------------------------------------------------------
// im2col -> f16 A-fragments. Each (m,k) activation element is gathered and
// converted to f16 exactly ONCE per layer (the GEMM pass then reuses it for
// every N tile group). Output layout is this code's WMMA A 16x32 operand order (same k order as B):
//   af[((mtile*Kt + ktile)*32 + lane)*16 + e]   (32B contiguous per lane)
// with e -> k_local = e + (lane>=16)*16, M row = lane&15.
// Per-block k decode: 32-entry LDS table (one decode per lane, no divisions
// in the gather loop).
// ---------------------------------------------------------------------------
__global__ void im2col_kernel(const float* __restrict__ in0, int c0,
                              const float* __restrict__ in1, int c1, int add_mode,
                              int H, int W, int K, int ks, int ksq, int pad, int dil,
                              _Float16* __restrict__ af) {
  __shared__ int2 tab[32];

  const int lane = threadIdx.x;
  const int half_id = lane >> 4;
  const int l15 = lane & 15;
  const int HW = H * W;
  const int M = BATCH * HW;
  const int kt = blockIdx.y << 5;

  // decode this block's 32 k values
  {
    int k = kt + lane;
    int off = 0, meta = -1;
    if (k < K) {
      int ci = k / ksq;
      int r = k - ci * ksq;
      int fy = r / ks;
      int fx = r - fy * ks;
      int dy = fy * dil - pad;
      int dx = fx * dil - pad;
      int sel = 0, cc = ci;
      if (!add_mode && ci >= c0) { sel = 1; cc = ci - c0; }
      off = cc * HW + dy * W + dx;
      meta = (dy + 8) | ((dx + 8) << 5) | (sel << 10);
    }
    tab[lane].x = off;
    tab[lane].y = meta;
  }
  __syncthreads();

  int m_global = blockIdx.x * 16 + l15;
  bool mvalid = m_global < M;
  int mg = mvalid ? m_global : 0;
  int b = mg / HW;
  int rem = mg - b * HW;
  int y = rem / W;
  int x = rem - y * W;

  const float* p0 = in0 + (size_t)b * c0 * HW + (size_t)y * W + x;
  const float* p1 = in1 ? (in1 + (size_t)b * c1 * HW + (size_t)y * W + x) : p0;

  v16h a;
#pragma unroll
  for (int e = 0; e < 16; ++e) {
    int kl = half_id * 16 + e;
    int2 te = tab[kl];
    float v = 0.f;
    if (te.y >= 0 && mvalid) {
      int dy = (te.y & 31) - 8;
      int dx = ((te.y >> 5) & 31) - 8;
      if ((unsigned)(y + dy) < (unsigned)H && (unsigned)(x + dx) < (unsigned)W) {
        if (add_mode) v = p0[te.x] + p1[te.x];
        else v = (te.y & (1 << 10)) ? p1[te.x] : p0[te.x];
      }
    }
    a[e] = (_Float16)v;
  }
  *(v16h*)(af + (((size_t)blockIdx.x * gridDim.y + blockIdx.y) * 32 + lane) * 16) = a;
}

// ---------------------------------------------------------------------------
// Pure-WMMA GEMM: one wave computes a 16M x 32N tile; inner loop is one
// contiguous A-fragment load, two contiguous B-fragment loads, and two
// V_WMMA_F32_16X16X32_F16 issues -- no address math, no conversions.
// D 16x16 f32: VGPR r -> M = r + 8*(lane>=16), N = lane&15.
// Epilogue fuses: bias, residual (concat / nearest-upsample), BN, lrelu/sigmoid.
// ---------------------------------------------------------------------------
__global__ void gemm_wmma_kernel(
    const _Float16* __restrict__ af, const _Float16* __restrict__ wfrag,
    int Kt, int Np, int H, int W,
    const float* __restrict__ bias, int Cout,
    const float* __restrict__ bn_s, const float* __restrict__ bn_t,
    int act,
    const float* __restrict__ res0, int rc0, const float* __restrict__ res1, int res_up,
    float* __restrict__ out) {
  const int lane = threadIdx.x;
  const int half_id = lane >> 4;
  const int l15 = lane & 15;
  const int HW = H * W;
  const int M = BATCH * HW;
  const int nbase = blockIdx.y * 32;

  v8f acc0 = {0.f, 0.f, 0.f, 0.f, 0.f, 0.f, 0.f, 0.f};
  v8f acc1 = {0.f, 0.f, 0.f, 0.f, 0.f, 0.f, 0.f, 0.f};

  const _Float16* ap = af + ((size_t)blockIdx.x * Kt * 32 + lane) * 16;
  const _Float16* wp = wfrag + (size_t)nbase * 32 + half_id * 16;

  for (int t = 0; t < Kt; ++t) {
    v16h afrag = *(const v16h*)(ap + (size_t)t * 512);
    const _Float16* wt = wp + (size_t)t * Np * 32;
    v16h bfrag0 = *(const v16h*)(wt + (size_t)l15 * 32);
    v16h bfrag1 = *(const v16h*)(wt + (size_t)(l15 + 16) * 32);
    acc0 = __builtin_amdgcn_wmma_f32_16x16x32_f16(false, afrag, false, bfrag0,
                                                  (short)0, acc0, false, false);
    acc1 = __builtin_amdgcn_wmma_f32_16x16x32_f16(false, afrag, false, bfrag1,
                                                  (short)0, acc1, false, false);
  }

#pragma unroll
  for (int nh = 0; nh < 2; ++nh) {
    int cout = nbase + nh * 16 + l15;
    if (cout >= Cout) continue;
    float bv = bias ? bias[cout] : 0.f;
    float bs = bn_s ? bn_s[cout] : 1.f;
    float bt = bn_t ? bn_t[cout] : 0.f;
#pragma unroll
    for (int r = 0; r < 8; ++r) {
      int mm = blockIdx.x * 16 + r + half_id * 8;
      if (mm >= M) continue;
      int bb = mm / HW;
      int rr = mm - bb * HW;
      int yy = rr / W;
      int xx = rr - yy * W;
      float v = (nh ? acc1[r] : acc0[r]) + bv;
      if (res0) {
        float rv;
        if (res_up) {
          rv = res0[(((size_t)bb * Cout + cout) * (H >> 1) + (yy >> 1)) * (W >> 1) + (xx >> 1)];
        } else if (cout < rc0) {
          rv = res0[(((size_t)bb * rc0 + cout) * H + yy) * W + xx];
        } else {
          rv = res1[(((size_t)bb * (Cout - rc0) + (cout - rc0)) * H + yy) * W + xx];
        }
        v += rv;
      }
      if (bn_s) v = v * bs + bt;
      if (act == 1) v = (v >= 0.f) ? v : 0.2f * v;
      else if (act == 2) v = 1.f / (1.f + __expf(-v));
      out[(((size_t)bb * Cout + cout) * H + yy) * W + xx] = v;
    }
  }
}

// ---------------------------------------------------------------------------
// Final dynamic filter at 384x384. Reads pre-pixel-shuffle head outputs:
//   weight chan (t=i*3+j) -> cw[ t*16 + sub ], offset (i,j,d) -> co[ (i*6+j*2+d)*16 + sub ]
// where sub = (y%4)*4 + (x%4), low-res pixel = (y/4, x/4).
// ---------------------------------------------------------------------------
__global__ void dynfilter_kernel(const float* __restrict__ depth,
                                 const float* __restrict__ cw,
                                 const float* __restrict__ co,
                                 float* __restrict__ out) {
  const int H = 384, W = 384, Hh = 96, Wh = 96;
  int i = blockIdx.x * blockDim.x + threadIdx.x;
  int tot = BATCH * H * W;
  if (i >= tot) return;
  int x = i % W;
  int y = (i / W) % H;
  int b = i / (W * H);
  int yh = y >> 2, ry = y & 3, xw = x >> 2, rx = x & 3;
  int sub = ry * 4 + rx;
  size_t plane = (size_t)Hh * Wh;
  size_t pix = (size_t)yh * Wh + xw;
  const float* cwb = cw + (size_t)b * 144 * plane;
  const float* cob = co + (size_t)b * 288 * plane;

  float wv[9];
  float mean = 0.f;
#pragma unroll
  for (int t = 0; t < 9; ++t) {
    wv[t] = cwb[(size_t)(t * 16 + sub) * plane + pix];
    mean += wv[t];
  }
  mean *= (1.f / 9.f);

  const float* dpb = depth + (size_t)b * H * W;
  float sum = 0.f;
#pragma unroll
  for (int i3 = 0; i3 < 3; ++i3) {
#pragma unroll
    for (int j3 = 0; j3 < 3; ++j3) {
      int t = i3 * 3 + j3;
      float ox = cob[(size_t)((i3 * 6 + j3 * 2 + 0) * 16 + sub) * plane + pix];
      float oy = cob[(size_t)((i3 * 6 + j3 * 2 + 1) * 16 + sub) * plane + pix];
      // fx = ((cx+1)*K-1)/2 with cx=(7+i3+ox)/K*2-1  ==>  fx = 7+i3+ox-0.5
      float fx = 6.5f + (float)i3 + ox;
      float fy = 6.5f + (float)j3 + oy;
      float x0f = floorf(fx), y0f = floorf(fy);
      float wx = fx - x0f, wy = fy - y0f;
      int x0 = (int)x0f, y0 = (int)y0f;
      float s = 0.f;
#pragma unroll
      for (int dy = 0; dy < 2; ++dy) {
#pragma unroll
        for (int dx = 0; dx < 2; ++dx) {
          int xi = x0 + dx, yi = y0 + dy;
          float wgt = (dx ? wx : 1.f - wx) * (dy ? wy : 1.f - wy);
          float val = 0.f;
          if (xi >= 0 && xi < 15 && yi >= 0 && yi < 15) {
            int gy = y + yi - 7, gx = x + xi - 7;
            if (gy >= 0 && gy < H && gx >= 0 && gx < W)
              val = dpb[(size_t)gy * W + gx];
          }
          s += val * wgt;
        }
      }
      sum += s * (wv[t] - mean);
    }
  }
  out[i] = dpb[(size_t)y * W + x] + sum;
}

// ---------------------------------------------------------------------------
// Host orchestration
// ---------------------------------------------------------------------------
struct WB {
  const _Float16* w;
  int K, Kp, Np;
};

extern "C" void kernel_launch(void* const* d_in, const int* in_sizes, int n_in,
                              void* d_out, int out_size, void* d_ws, size_t ws_size,
                              hipStream_t stream) {
  (void)in_sizes; (void)n_in; (void)out_size; (void)ws_size;
  const float* image = (const float*)d_in[0];
  const float* depth = (const float*)d_in[1];
  auto P = [&](int i) { return (const float*)d_in[i]; };

  char* ws = (char*)d_ws;
  size_t off = 0;
  auto alloc = [&](size_t bytes) -> void* {
    void* p = ws + off;
    off += (bytes + 255) & ~(size_t)255;
    return p;
  };

  auto prepw = [&](const float* src, int Cout, int Cin, int ks) -> WB {
    int K = Cin * ks * ks;
    int Kp = (K + 31) & ~31;
    int Np = (Cout + 31) & ~31;  // 32-wide N tiles per wave
    _Float16* dst = (_Float16*)alloc((size_t)Kp * Np * sizeof(_Float16));
    int tot = Kp * Np;
    prep_weight_kernel<<<dim3((tot + 255) / 256), dim3(256), 0, stream>>>(src, dst, K, Kp, Np, Cout);
    WB r; r.w = dst; r.K = K; r.Kp = Kp; r.Np = Np;
    return r;
  };

  auto prepbn = [&](int base, int C, float** s, float** t) {
    *s = (float*)alloc((size_t)C * 4);
    *t = (float*)alloc((size_t)C * 4);
    prep_bn_kernel<<<dim3((C + 127) / 128), dim3(128), 0, stream>>>(P(base), P(base + 1), P(base + 2),
                                                                    P(base + 3), C, *s, *t);
  };

  // A-fragment staging buffer: max M = 2*96*96 = 18432 (1152 tiles), max Kp = 960
  // -> 1152 * 30 * 512 halfs = 35.4 MB (L2-resident)
  _Float16* afbuf = (_Float16*)alloc((size_t)1152 * 30 * 512 * sizeof(_Float16));

  auto conv = [&](WB wb, const float* in0, int c0, const float* in1, int c1, int addm,
                  int H, int W, const float* bias, int Cout, int ks, int pad, int dil,
                  const float* bns, const float* bnt, int act,
                  const float* r0, int rc0, const float* r1, int rup, float* outp) {
    int M = BATCH * H * W;
    int Mt = (M + 15) / 16;
    int Kt = wb.Kp >> 5;
    im2col_kernel<<<dim3(Mt, Kt), dim3(32), 0, stream>>>(in0, c0, in1, c1, addm, H, W, wb.K, ks,
                                                         ks * ks, pad, dil, afbuf);
    gemm_wmma_kernel<<<dim3(Mt, wb.Np / 32), dim3(32), 0, stream>>>(afbuf, wb.w, Kt, wb.Np, H, W,
                                                                    bias, Cout, bns, bnt, act, r0,
                                                                    rc0, r1, rup, outp);
  };

  // ---- weights (param leaf order: insertion order of make_params dict) ----
  WB w_dp1 = prepw(P(2), 32, 16, 3);
  WB w_rgb1 = prepw(P(4), 32, 16, 3);
  WB m1w1 = prepw(P(6), 32, 32, 3), m1w2 = prepw(P(8), 32, 32, 3), m1w4 = prepw(P(10), 32, 32, 1);
  WB m2w1 = prepw(P(12), 56, 56, 3), m2w2 = prepw(P(14), 56, 56, 3), m2w4 = prepw(P(16), 56, 56, 1);
  WB m3w1 = prepw(P(18), 80, 80, 3), m3w2 = prepw(P(20), 80, 80, 3), m3w4 = prepw(P(22), 80, 80, 1);
  WB m4w1 = prepw(P(24), 104, 104, 3), m4w2 = prepw(P(26), 104, 104, 3), m4w4 = prepw(P(28), 104, 104, 1);
  WB c2h2h = prepw(P(30), 24, 32, 3), c2h2l = prepw(P(31), 8, 32, 3);
  WB c3h2h = prepw(P(40), 24, 24, 3), c3l2h = prepw(P(41), 24, 8, 3);
  WB c3h2l = prepw(P(42), 8, 24, 3), c3l2l = prepw(P(43), 8, 8, 3);
  WB c4h2h = prepw(P(52), 24, 24, 3), c4l2h = prepw(P(53), 24, 8, 3);
  WB wconv1 = prepw(P(64), 128, 104, 3);
  WB wcw = prepw(P(66), 144, 128, 1);
  WB wco = prepw(P(68), 288, 128, 1);

  float *bn2hs, *bn2ht, *bn2ls, *bn2lt, *bn3hs, *bn3ht, *bn3ls, *bn3lt, *bn4hs, *bn4ht;
  prepbn(32, 24, &bn2hs, &bn2ht);
  prepbn(36, 8, &bn2ls, &bn2lt);
  prepbn(44, 24, &bn3hs, &bn3ht);
  prepbn(48, 8, &bn3ls, &bn3lt);
  prepbn(56, 24, &bn4hs, &bn4ht);

  // ---- activations ----
  const int h = 96, w = 96, hl = 48, wl = 48;
  const size_t HW = (size_t)h * w, HL = (size_t)hl * wl;
  float* re_im = (float*)alloc(BATCH * 16 * HW * 4);
  float* re_dp = (float*)alloc(BATCH * 16 * HW * 4);
  float* dp_in = (float*)alloc(BATCH * 32 * HW * 4);
  float* t1 = (float*)alloc(BATCH * 104 * HW * 4);
  float* t2 = (float*)alloc(BATCH * 104 * HW * 4);
  float* dp1 = (float*)alloc(BATCH * 32 * HW * 4);
  float* rgb1 = (float*)alloc(BATCH * 32 * HW * 4);
  float* h2 = (float*)alloc(BATCH * 24 * HW * 4);
  float* l2 = (float*)alloc(BATCH * 8 * HL * 4);
  float* dp2 = (float*)alloc(BATCH * 56 * HW * 4);
  float* h3 = (float*)alloc(BATCH * 24 * HW * 4);
  float* l3 = (float*)alloc(BATCH * 8 * HL * 4);
  float* dp3 = (float*)alloc(BATCH * 80 * HW * 4);
  float* h4 = (float*)alloc(BATCH * 24 * HW * 4);
  float* dp4 = (float*)alloc(BATCH * 104 * HW * 4);
  float* xconv1 = (float*)alloc(BATCH * 128 * HW * 4);
  float* cwout = (float*)alloc(BATCH * 144 * HW * 4);
  float* coout = (float*)alloc(BATCH * 288 * HW * 4);
  float* pool = (float*)alloc(BATCH * 32 * HL * 4);
  float* ltmp = (float*)alloc(BATCH * 24 * HL * 4);
  float* lt2 = (float*)alloc(BATCH * 8 * HL * 4);

  const float* NF = nullptr;

  // ---- forward ----
  {
    int tot = BATCH * 16 * h * w;
    resample_kernel<<<dim3((tot + 255) / 256), dim3(256), 0, stream>>>(image, 3, re_im, h, w);
    resample_kernel<<<dim3((tot + 255) / 256), dim3(256), 0, stream>>>(depth, 1, re_dp, h, w);
  }

  conv(w_dp1, re_dp, 16, NF, 0, 0, h, w, P(3), 32, 3, 1, 1, NF, NF, 1, NF, 0, NF, 0, dp_in);
  // msb1
  conv(m1w1, dp_in, 32, NF, 0, 0, h, w, P(7), 32, 3, 1, 1, NF, NF, 1, NF, 0, NF, 0, t1);
  conv(m1w2, dp_in, 32, NF, 0, 0, h, w, P(9), 32, 3, 2, 2, NF, NF, 1, NF, 0, NF, 0, t2);
  conv(m1w4, t1, 32, t2, 32, 1, h, w, P(11), 32, 1, 0, 1, NF, NF, 0, dp_in, 32, NF, 0, dp1);

  conv(w_rgb1, re_im, 16, NF, 0, 0, h, w, P(5), 32, 3, 1, 1, NF, NF, 1, NF, 0, NF, 0, rgb1);
  // cbl2 (oct_first)
  conv(c2h2h, rgb1, 32, NF, 0, 0, h, w, NF, 24, 3, 1, 1, bn2hs, bn2ht, 1, NF, 0, NF, 0, h2);
  {
    int tot = BATCH * 32 * hl * wl;
    avgpool2_kernel<<<dim3((tot + 255) / 256), dim3(256), 0, stream>>>(rgb1, pool, 32, hl, wl);
  }
  conv(c2h2l, pool, 32, NF, 0, 0, hl, wl, NF, 8, 3, 1, 1, bn2ls, bn2lt, 1, NF, 0, NF, 0, l2);

  // msb2 on concat(dp1,h2)
  conv(m2w1, dp1, 32, h2, 24, 0, h, w, P(13), 56, 3, 1, 1, NF, NF, 1, NF, 0, NF, 0, t1);
  conv(m2w2, dp1, 32, h2, 24, 0, h, w, P(15), 56, 3, 2, 2, NF, NF, 1, NF, 0, NF, 0, t2);
  conv(m2w4, t1, 56, t2, 56, 1, h, w, P(17), 56, 1, 0, 1, NF, NF, 0, dp1, 32, h2, 0, dp2);

  // cbl3 (oct_mid): h3 = lrelu(bn(conv(h2) + up(conv(l2)))), l3 = lrelu(bn(conv(l2) + conv(pool(h2))))
  conv(c3l2h, l2, 8, NF, 0, 0, hl, wl, NF, 24, 3, 1, 1, NF, NF, 0, NF, 0, NF, 0, ltmp);
  conv(c3h2h, h2, 24, NF, 0, 0, h, w, NF, 24, 3, 1, 1, bn3hs, bn3ht, 1, ltmp, 24, NF, 1, h3);
  {
    int tot = BATCH * 24 * hl * wl;
    avgpool2_kernel<<<dim3((tot + 255) / 256), dim3(256), 0, stream>>>(h2, pool, 24, hl, wl);
  }
  conv(c3h2l, pool, 24, NF, 0, 0, hl, wl, NF, 8, 3, 1, 1, NF, NF, 0, NF, 0, NF, 0, lt2);
  conv(c3l2l, l2, 8, NF, 0, 0, hl, wl, NF, 8, 3, 1, 1, bn3ls, bn3lt, 1, lt2, 8, NF, 0, l3);

  // msb3 on concat(dp2,h3)
  conv(m3w1, dp2, 56, h3, 24, 0, h, w, P(19), 80, 3, 1, 1, NF, NF, 1, NF, 0, NF, 0, t1);
  conv(m3w2, dp2, 56, h3, 24, 0, h, w, P(21), 80, 3, 2, 2, NF, NF, 1, NF, 0, NF, 0, t2);
  conv(m3w4, t1, 80, t2, 80, 1, h, w, P(23), 80, 1, 0, 1, NF, NF, 0, dp2, 56, h3, 0, dp3);

  // cbl4: only the high branch is consumed downstream (l4 is dead in the reference)
  conv(c4l2h, l3, 8, NF, 0, 0, hl, wl, NF, 24, 3, 1, 1, NF, NF, 0, NF, 0, NF, 0, ltmp);
  conv(c4h2h, h3, 24, NF, 0, 0, h, w, NF, 24, 3, 1, 1, bn4hs, bn4ht, 1, ltmp, 24, NF, 1, h4);

  // msb4 on concat(dp3,h4)
  conv(m4w1, dp3, 80, h4, 24, 0, h, w, P(25), 104, 3, 1, 1, NF, NF, 1, NF, 0, NF, 0, t1);
  conv(m4w2, dp3, 80, h4, 24, 0, h, w, P(27), 104, 3, 2, 2, NF, NF, 1, NF, 0, NF, 0, t2);
  conv(m4w4, t1, 104, t2, 104, 1, h, w, P(29), 104, 1, 0, 1, NF, NF, 0, dp3, 80, h4, 0, dp4);

  // heads
  conv(wconv1, dp4, 104, NF, 0, 0, h, w, P(65), 128, 3, 1, 1, NF, NF, 0, NF, 0, NF, 0, xconv1);
  conv(wcw, xconv1, 128, NF, 0, 0, h, w, P(67), 144, 1, 0, 1, NF, NF, 2, NF, 0, NF, 0, cwout);
  conv(wco, xconv1, 128, NF, 0, 0, h, w, P(69), 288, 1, 0, 1, NF, NF, 2, NF, 0, NF, 0, coout);

  // dynamic filtering at full resolution
  {
    int tot = BATCH * 384 * 384;
    dynfilter_kernel<<<dim3((tot + 255) / 256), dim3(256), 0, stream>>>(depth, cwout, coout,
                                                                        (float*)d_out);
  }
}
